// Position_attention_20744692040369
// MI455X (gfx1250) — matrix operand order repaired and checked by reading, not compiled, not run on hardware
//
#include <hip/hip_runtime.h>
#include <hip/hip_bf16.h>

// Problem dims (from reference): B=4, D=H=W=32, C=256
#define BATCH   4
#define NSP     32768      // D*H*W
#define CH      256
#define CH3     768        // 3*C (packed Wq|Wk|Wv columns)
#define KSPLIT  16         // split-K factor for the energy GEMM
#define LDW     72         // padded LDS row width (halves) to spread banks

typedef __attribute__((ext_vector_type(16))) __bf16 v16bf;
typedef __attribute__((ext_vector_type(2)))  __bf16 v2bf;
typedef __attribute__((ext_vector_type(8)))  float  v8f;

union FragU { uint4 u[2]; v16bf v; };

// f32 -> bf16 via compiler cast (backend picks native cvt if present)
__device__ __forceinline__ unsigned short f32_to_bf16(float f) {
  __bf16 h = (__bf16)f;
  return __builtin_bit_cast(unsigned short, h);
}
__device__ __forceinline__ unsigned cvt2(float a, float b) {
  v2bf p;
  p[0] = (__bf16)a;
  p[1] = (__bf16)b;
  return __builtin_bit_cast(unsigned, p);
}

// ---- CDNA5 async copy: global -> LDS, tracked by ASYNCcnt ---------------
__device__ __forceinline__ void async_b128(unsigned short* lds_dst, const unsigned short* gsrc) {
  unsigned loff = (unsigned)(size_t)lds_dst;   // LDS aperture: low 32 bits are the LDS byte address
  asm volatile("global_load_async_to_lds_b128 %0, %1, off"
               :: "v"(loff), "v"(gsrc) : "memory");
}
__device__ __forceinline__ void wait_async_le4() {
  asm volatile("s_wait_asynccnt 0x4" ::: "memory");
}
__device__ __forceinline__ void wait_async_0() {
  asm volatile("s_wait_asynccnt 0x0" ::: "memory");
}

// ---- WMMA fragment loads from LDS (CDNA5 layouts, wave32) ----------------
// A (16x32 bf16): lanes 0-15: row M=L, K chunks [0..7],[16..23];
//                 lanes 16-31: row M=L-16, K chunks [8..15],[24..31].
__device__ __forceinline__ v16bf a_frag(const unsigned short (*A)[LDW], int wy, int lane, int kc) {
  int row  = 16 * wy + (lane & 15);
  int base = kc * 32 + (lane >> 4) * 8;
  FragU r;
  r.u[0] = *(const uint4*)&A[row][base];
  r.u[1] = *(const uint4*)&A[row][base + 16];
  return r.v;
}
// B (32x16 bf16) held transposed in LDS as Bt[n][k]:
// lanes 0-15: col N=L, K chunks [0..7],[16..23]; lanes 16-31: col N=L-16, K chunks [8..15],[24..31].
__device__ __forceinline__ v16bf b_frag(const unsigned short (*Bt)[LDW], int ncol0, int lane, int kc) {
  int row  = ncol0 + (lane & 15);
  int base = kc * 32 + (lane >> 4) * 8;
  FragU r;
  r.u[0] = *(const uint4*)&Bt[row][base];
  r.u[1] = *(const uint4*)&Bt[row][base + 16];
  return r.v;
}
// One K=64 stage: 2 k-chunks x 2 N-subtiles = 4 WMMAs per wave.
__device__ __forceinline__ void mma_stage(const unsigned short (*Al)[LDW],
                                          const unsigned short (*Bl)[LDW],
                                          int wy, int wx, int lane, v8f* acc) {
  #pragma unroll
  for (int kc = 0; kc < 2; ++kc) {
    v16bf a  = a_frag(Al, wy, lane, kc);
    v16bf b0 = b_frag(Bl, 32 * wx,      lane, kc);
    v16bf b1 = b_frag(Bl, 32 * wx + 16, lane, kc);
    acc[0] = __builtin_amdgcn_wmma_f32_16x16x32_bf16(false, a, false, b0, (short)0, acc[0], false, false);
    acc[1] = __builtin_amdgcn_wmma_f32_16x16x32_bf16(false, a, false, b1, (short)0, acc[1], false, false);
  }
}

// ---- Kernel 0: transpose+convert weights to bf16 Wt[j][k] = W_sel[k][j&255]
__global__ __launch_bounds__(256) void prep_kernel(
    const float* __restrict__ Wq, const float* __restrict__ Wk, const float* __restrict__ Wv,
    const float* __restrict__ bq, const float* __restrict__ bk, const float* __restrict__ bv,
    unsigned short* __restrict__ Wt, float* __restrict__ bias_all) {
  int j = blockIdx.x, k = threadIdx.x;
  int sel = j >> 8, c = j & 255;
  const float* W = (sel == 0) ? Wq : ((sel == 1) ? Wk : Wv);
  Wt[(size_t)j * CH + k] = f32_to_bf16(W[(size_t)k * CH + c]);
  if (k == 0) {
    const float* bb = (sel == 0) ? bq : ((sel == 1) ? bk : bv);
    bias_all[j] = bb[c];
  }
}

// ---- Kernel 1: QKV projection; Qt/Kt stored transposed [b][c][n], V [b][n][c]
__global__ __launch_bounds__(256) void qkv_kernel(
    const float* __restrict__ x, const unsigned short* __restrict__ Wt,
    const float* __restrict__ bias_all,
    unsigned short* __restrict__ Qt, unsigned short* __restrict__ Kt,
    unsigned short* __restrict__ V) {
  __shared__ __align__(16) unsigned short Alds[64][LDW];
  __shared__ __align__(16) unsigned short Blds[64][LDW];
  const int b  = blockIdx.z;
  const int n0 = blockIdx.x * 64;     // spatial rows (M)
  const int j0 = blockIdx.y * 64;     // output columns (N of 768)
  const int t = threadIdx.x, lane = t & 31, wave = t >> 5;
  const int wy = wave >> 1, wx = wave & 1;
  const int srow = t >> 2, scol = (t & 3) * 16;
  const float* xb = x + (size_t)b * NSP * CH;
  v8f z = {0.f,0.f,0.f,0.f,0.f,0.f,0.f,0.f};
  v8f acc[2] = {z, z};

  for (int k0 = 0; k0 < CH; k0 += 64) {
    { // stage A: 64x64 f32 -> bf16 (packed convert), 32B per thread
      const float* src = xb + (size_t)(n0 + srow) * CH + (k0 + scol);
      float4 f0 = *(const float4*)(src);
      float4 f1 = *(const float4*)(src + 4);
      float4 f2 = *(const float4*)(src + 8);
      float4 f3 = *(const float4*)(src + 12);
      uint4 p, q;
      p.x = cvt2(f0.x, f0.y); p.y = cvt2(f0.z, f0.w);
      p.z = cvt2(f1.x, f1.y); p.w = cvt2(f1.z, f1.w);
      q.x = cvt2(f2.x, f2.y); q.y = cvt2(f2.z, f2.w);
      q.z = cvt2(f3.x, f3.y); q.w = cvt2(f3.z, f3.w);
      *(uint4*)&Alds[srow][scol]     = p;
      *(uint4*)&Alds[srow][scol + 8] = q;
    }
    { // stage B (Wt already transposed): 64 rows x 64 k bf16
      const unsigned short* src = Wt + (size_t)(j0 + srow) * CH + (k0 + scol);
      *(uint4*)&Blds[srow][scol]     = *(const uint4*)(src);
      *(uint4*)&Blds[srow][scol + 8] = *(const uint4*)(src + 8);
    }
    __syncthreads();
    mma_stage(Alds, Blds, wy, wx, lane, acc);
    __syncthreads();
  }

  // Epilogue. D layout: lane value r sits at (M = r + 8*(lane>=16), Nloc = lane&15).
  const int hi = lane >> 4;
  const int nbase = n0 + 16 * wy + 8 * hi;         // 8 consecutive spatial rows
  #pragma unroll
  for (int sub = 0; sub < 2; ++sub) {
    const int colg = j0 + 32 * wx + 16 * sub + (lane & 15);   // 0..767
    const float bv = bias_all[colg];
    v8f a = acc[sub];
    if (colg < 512) {  // Q or K: transposed store -> one b128 per lane
      unsigned short* dst = (colg < 256) ? Qt : Kt;
      const int c = colg & 255;
      uint4 p;
      p.x = cvt2(a[0] + bv, a[1] + bv);
      p.y = cvt2(a[2] + bv, a[3] + bv);
      p.z = cvt2(a[4] + bv, a[5] + bv);
      p.w = cvt2(a[6] + bv, a[7] + bv);
      *(uint4*)(dst + ((size_t)b * CH + c) * NSP + nbase) = p;
    } else {           // V: natural [n][c] store
      const int c = colg - 512;
      #pragma unroll
      for (int r = 0; r < 8; ++r)
        V[((size_t)b * NSP + nbase + r) * CH + c] = f32_to_bf16(a[r] + bv);
    }
  }
}

// ---- Kernel 2: energy partials E[b][m][n] = sum_k Qt[b][m][k]*Kt[b][n][k]
// Async double-buffered staging (ASYNCcnt), deterministic split-K partials.
__global__ __launch_bounds__(256) void energy_kernel(
    const unsigned short* __restrict__ Qt, const unsigned short* __restrict__ Kt,
    float* __restrict__ Epart) {
  __shared__ __align__(16) unsigned short Alds[2][64][LDW];
  __shared__ __align__(16) unsigned short Blds[2][64][LDW];
  const int b = blockIdx.z / KSPLIT, s = blockIdx.z % KSPLIT;
  const int m0 = blockIdx.y * 64, n0 = blockIdx.x * 64;
  const int t = threadIdx.x, lane = t & 31, wave = t >> 5;
  const int wy = wave >> 1, wx = wave & 1;
  const int srow = t >> 2, scol = (t & 3) * 16;
  const int kbeg = s * (NSP / KSPLIT);
  const int NSTAGES = (NSP / KSPLIT) / 64;   // 32
  const unsigned short* Ag = Qt + ((size_t)b * CH + m0 + srow) * NSP + kbeg + scol;
  const unsigned short* Bg = Kt + ((size_t)b * CH + n0 + srow) * NSP + kbeg + scol;
  v8f z = {0.f,0.f,0.f,0.f,0.f,0.f,0.f,0.f};
  v8f acc[2] = {z, z};

  // prologue: stages 0 and 1 in flight
  #pragma unroll
  for (int p = 0; p < 2; ++p) {
    async_b128(&Alds[p][srow][scol],     Ag + p * 64);
    async_b128(&Alds[p][srow][scol + 8], Ag + p * 64 + 8);
    async_b128(&Blds[p][srow][scol],     Bg + p * 64);
    async_b128(&Blds[p][srow][scol + 8], Bg + p * 64 + 8);
  }
  for (int i = 0; i < NSTAGES; ++i) {
    if (i + 1 < NSTAGES) wait_async_le4(); else wait_async_0();  // stage i landed
    __syncthreads();
    mma_stage(Alds[i & 1], Blds[i & 1], wy, wx, lane, acc);
    __syncthreads();                                            // all reads of buf done
    if (i + 2 < NSTAGES) {
      const int k = (i + 2) * 64;
      async_b128(&Alds[i & 1][srow][scol],     Ag + k);
      async_b128(&Alds[i & 1][srow][scol + 8], Ag + k + 8);
      async_b128(&Blds[i & 1][srow][scol],     Bg + k);
      async_b128(&Blds[i & 1][srow][scol + 8], Bg + k + 8);
    }
  }

  const int mbase = m0 + 16 * wy + 8 * (lane >> 4);
  #pragma unroll
  for (int sub = 0; sub < 2; ++sub) {
    const int ncol = n0 + 32 * wx + 16 * sub + (lane & 15);
    float* dst = Epart + (((size_t)s * BATCH + b) * CH + mbase) * CH + ncol;
    #pragma unroll
    for (int r = 0; r < 8; ++r) dst[(size_t)r * CH] = acc[sub][r];
  }
}

// ---- Kernel 3: deterministic split-K reduction ---------------------------
__global__ __launch_bounds__(256) void reduce_kernel(
    const float* __restrict__ Epart, float* __restrict__ energy) {
  size_t idx = (size_t)blockIdx.x * 256 + threadIdx.x;   // over B*C*C
  const size_t stride = (size_t)BATCH * CH * CH;
  float e = 0.f;
  #pragma unroll
  for (int s = 0; s < KSPLIT; ++s) e += Epart[s * stride + idx];
  energy[idx] = e;
}

// ---- Kernel 4: row softmax over C=256; attn stored bf16 [b][f][c] --------
__global__ __launch_bounds__(256) void softmax_kernel(
    const float* __restrict__ energy, unsigned short* __restrict__ attn) {
  const int row = blockIdx.x;          // b*256 + f
  const int t = threadIdx.x;
  __shared__ float redA[8], redB[8];
  float e = energy[(size_t)row * CH + t];
  float m = e;
  #pragma unroll
  for (int off = 16; off > 0; off >>= 1) m = fmaxf(m, __shfl_xor(m, off, 32));
  if ((t & 31) == 0) redA[t >> 5] = m;
  __syncthreads();
  float mx = redA[0];
  #pragma unroll
  for (int i = 1; i < 8; ++i) mx = fmaxf(mx, redA[i]);
  float p = __expf(e - mx);
  float s = p;
  #pragma unroll
  for (int off = 16; off > 0; off >>= 1) s += __shfl_xor(s, off, 32);
  if ((t & 31) == 0) redB[t >> 5] = s;
  __syncthreads();
  float tot = redB[0];
  #pragma unroll
  for (int i = 1; i < 8; ++i) tot += redB[i];
  attn[(size_t)row * CH + t] = f32_to_bf16(p / tot);
}

// ---- Kernel 5: outT[f][n] = sum_c attn[f][c]*V[n][c];
//      d_out[b][n][f] = gamma*outT + x[b][n][f].  Async double-buffered. ---
__global__ __launch_bounds__(256) void out_kernel(
    const unsigned short* __restrict__ attn, const unsigned short* __restrict__ V,
    const float* __restrict__ x, const float* __restrict__ gamma,
    float* __restrict__ out) {
  __shared__ __align__(16) unsigned short Alds[2][64][LDW];
  __shared__ __align__(16) unsigned short Blds[2][64][LDW];
  const int b = blockIdx.z;
  const int f0 = blockIdx.y * 64;      // M dim: output channel f
  const int n0 = blockIdx.x * 64;      // N dim: spatial position
  const int t = threadIdx.x, lane = t & 31, wave = t >> 5;
  const int wy = wave >> 1, wx = wave & 1;
  const int srow = t >> 2, scol = (t & 3) * 16;
  const int NSTAGES = CH / 64;         // 4
  const unsigned short* Ag = attn + ((size_t)b * CH  + f0 + srow) * CH + scol;
  const unsigned short* Bg = V    + ((size_t)b * NSP + n0 + srow) * CH + scol;
  v8f z = {0.f,0.f,0.f,0.f,0.f,0.f,0.f,0.f};
  v8f acc[2] = {z, z};

  #pragma unroll
  for (int p = 0; p < 2; ++p) {
    async_b128(&Alds[p][srow][scol],     Ag + p * 64);
    async_b128(&Alds[p][srow][scol + 8], Ag + p * 64 + 8);
    async_b128(&Blds[p][srow][scol],     Bg + p * 64);
    async_b128(&Blds[p][srow][scol + 8], Bg + p * 64 + 8);
  }
  for (int i = 0; i < NSTAGES; ++i) {
    if (i + 1 < NSTAGES) wait_async_le4(); else wait_async_0();
    __syncthreads();
    mma_stage(Alds[i & 1], Blds[i & 1], wy, wx, lane, acc);
    __syncthreads();
    if (i + 2 < NSTAGES) {
      const int k = (i + 2) * 64;
      async_b128(&Alds[i & 1][srow][scol],     Ag + k);
      async_b128(&Alds[i & 1][srow][scol + 8], Ag + k + 8);
      async_b128(&Blds[i & 1][srow][scol],     Bg + k);
      async_b128(&Blds[i & 1][srow][scol + 8], Bg + k + 8);
    }
  }

  const int fb = f0 + 16 * wy + 8 * (lane >> 4);   // 8 consecutive channels
  const float g = gamma[0];
  #pragma unroll
  for (int sub = 0; sub < 2; ++sub) {
    const int ng = n0 + 32 * wx + 16 * sub + (lane & 15);
    const size_t off = ((size_t)b * NSP + ng) * CH + fb;
    float4 x0 = *(const float4*)(x + off);
    float4 x1 = *(const float4*)(x + off + 4);
    float4 o0, o1;
    o0.x = g * acc[sub][0] + x0.x; o0.y = g * acc[sub][1] + x0.y;
    o0.z = g * acc[sub][2] + x0.z; o0.w = g * acc[sub][3] + x0.w;
    o1.x = g * acc[sub][4] + x1.x; o1.y = g * acc[sub][5] + x1.y;
    o1.z = g * acc[sub][6] + x1.z; o1.w = g * acc[sub][7] + x1.w;
    *(float4*)(out + off)     = o0;
    *(float4*)(out + off + 4) = o1;
  }
}

// ---- Workspace layout (bytes, 256-aligned) -------------------------------
#define WT_OFF     ((size_t)0)                     // 768*256*2   = 393216
#define BIAS_OFF   ((size_t)393216)                // 768*4       = 3072
#define QT_OFF     ((size_t)396288)                // 4*256*32768*2 = 67108864
#define KT_OFF     ((size_t)67505152)
#define V_OFF      ((size_t)134614016)
#define EPART_OFF  ((size_t)201722880)             // 16*4*256*256*4 = 16777216
#define ENERGY_OFF ((size_t)218500096)             // 4*256*256*4 = 1048576
#define ATTN_OFF   ((size_t)219548672)             // 4*256*256*2 = 524288

extern "C" void kernel_launch(void* const* d_in, const int* in_sizes, int n_in,
                              void* d_out, int out_size, void* d_ws, size_t ws_size,
                              hipStream_t stream) {
  const float* x     = (const float*)d_in[0];
  const float* Wq    = (const float*)d_in[1];
  const float* bq    = (const float*)d_in[2];
  const float* Wk    = (const float*)d_in[3];
  const float* bk    = (const float*)d_in[4];
  const float* Wv    = (const float*)d_in[5];
  const float* bv    = (const float*)d_in[6];
  const float* gamma = (const float*)d_in[7];
  char* ws = (char*)d_ws;
  unsigned short* Wt   = (unsigned short*)(ws + WT_OFF);
  float*          bias = (float*)(ws + BIAS_OFF);
  unsigned short* Qt   = (unsigned short*)(ws + QT_OFF);
  unsigned short* Kt   = (unsigned short*)(ws + KT_OFF);
  unsigned short* V    = (unsigned short*)(ws + V_OFF);
  float*          Epart  = (float*)(ws + EPART_OFF);
  float*          energy = (float*)(ws + ENERGY_OFF);
  unsigned short* attn   = (unsigned short*)(ws + ATTN_OFF);

  prep_kernel<<<dim3(CH3), 256, 0, stream>>>(Wq, Wk, Wv, bq, bk, bv, Wt, bias);
  qkv_kernel<<<dim3(NSP / 64, CH3 / 64, BATCH), 256, 0, stream>>>(x, Wt, bias, Qt, Kt, V);
  energy_kernel<<<dim3(CH / 64, CH / 64, BATCH * KSPLIT), 256, 0, stream>>>(Qt, Kt, Epart);
  reduce_kernel<<<dim3((BATCH * CH * CH) / 256), 256, 0, stream>>>(Epart, energy);
  softmax_kernel<<<dim3(BATCH * CH), 256, 0, stream>>>(energy, attn);
  out_kernel<<<dim3(NSP / 64, CH / 64, BATCH), 256, 0, stream>>>(attn, V, x, gamma, (float*)d_out);
}
